// OnlineNcdeAligner_52699248722469
// MI455X (gfx1250) — matrix operand order repaired and checked by reading, not compiled, not run on hardware
//
#include <hip/hip_runtime.h>
#include <hip/hip_bf16.h>
#include <math.h>

// ---------------------------------------------------------------------------
// OnlineNcdeAligner on MI455X (gfx1250): all convolutions lowered to
// implicit-im2col GEMMs through v_wmma_f32_16x16x32_f16 (wave32 WMMA).
// ---------------------------------------------------------------------------

typedef __attribute__((ext_vector_type(16))) _Float16 v16h;
typedef __attribute__((ext_vector_type(8)))  float    v8f;

constexpr int kXD   = 100;          // downsampled X
constexpr int kYD   = 100;          // downsampled Y
constexpr int kZD   = 16;           // Z
constexpr int kNCOL = kXD * kYD;    // 10000 z-columns
constexpr int kNPOS = kNCOL * kZD;  // 160000 voxels
constexpr int kCF   = 64;           // feature channels F == H
constexpr int kCI   = 32;           // INNER
constexpr int kNCLS = 18;           // logit classes
constexpr int kXF   = 200;          // full-res X
constexpr int kYF   = 200;          // full-res Y

// ---- WMMA fragment helpers (contiguous k order, same for A and B) -----

// A matrix 16x32 f16: lane l<16 -> row M=l, K = {0..15};
// lanes 16..31 -> row M=l-16, K = {16..31}.
__device__ __forceinline__ v16h frag_a_lds(const _Float16* tile, int lane) {
  const int r  = lane & 15;
  const int kb = (lane < 16) ? 0 : 16;
  v16h a;
#pragma unroll
  for (int i = 0; i < 16; ++i) {
    a[i]     = tile[r * 32 + kb + i];
  }
  return a;
}

// B matrix 32x16 f16 (stored as w16[N][Kpad]): lane l holds column N=l&15,
// K = (l<16 ? 0 : 16) + {0..15} (16 contiguous halves -> b128 loads).
__device__ __forceinline__ v16h frag_b_glb(const _Float16* __restrict__ w,
                                           int n0, int Kpad, int k0, int lane) {
  const _Float16* p = w + (size_t)(n0 + (lane & 15)) * Kpad + k0 + ((lane < 16) ? 0 : 16);
  v16h b;
#pragma unroll
  for (int i = 0; i < 16; ++i) b[i] = p[i];
  return b;
}

#define WMMA_F16(A, B, C) \
  __builtin_amdgcn_wmma_f32_16x16x32_f16(false, (A), false, (B), (short)0, (C), false, false)

// ---- weight packing: f32 (O,I,kx,ky,kz) -> f16 [Npad][Kpad], k=((ci*KX+kx)*KY+ky)*KZ+kz
__global__ void pack_w_kernel(const float* __restrict__ w, _Float16* __restrict__ w16,
                              int O, int I, int KX, int KY, int KZ, int Npad, int Kpad) {
  int idx = blockIdx.x * blockDim.x + threadIdx.x;
  if (idx >= Npad * Kpad) return;
  int n = idx / Kpad, k = idx % Kpad;
  float v = 0.f;
  int kvol = KX * KY * KZ;
  if (n < O && k < I * kvol) {
    int ci = k / kvol, r = k % kvol;
    int kx = r / (KY * KZ), ky = (r / KZ) % KY, kz = r % KZ;
    v = w[(((size_t)(n * I + ci) * KX + kx) * KY + ky) * KZ + kz];
  }
  w16[idx] = (_Float16)v;
}

// ---- backward-warp matrices A_k = inv(P[k]) @ P[k+1] (Gauss-Jordan) -------
__global__ void pose_mats_kernel(const float* __restrict__ poses, float* __restrict__ mats) {
  int k = threadIdx.x;
  if (k >= 3) return;
  float a[4][8];
  for (int r = 0; r < 4; ++r)
    for (int c = 0; c < 4; ++c) {
      a[r][c] = poses[k * 16 + r * 4 + c];
      a[r][4 + c] = (r == c) ? 1.f : 0.f;
    }
  for (int col = 0; col < 4; ++col) {
    int piv = col; float best = fabsf(a[col][col]);
    for (int r = col + 1; r < 4; ++r) { float t = fabsf(a[r][col]); if (t > best) { best = t; piv = r; } }
    if (piv != col)
      for (int c = 0; c < 8; ++c) { float t = a[col][c]; a[col][c] = a[piv][c]; a[piv][c] = t; }
    float inv = 1.f / a[col][col];
    for (int c = 0; c < 8; ++c) a[col][c] *= inv;
    for (int r = 0; r < 4; ++r) if (r != col) {
      float f = a[r][col];
      for (int c = 0; c < 8; ++c) a[r][c] -= f * a[col][c];
    }
  }
  for (int r = 0; r < 4; ++r)
    for (int c = 0; c < 4; ++c) {
      float s = 0.f;
      for (int j = 0; j < 4; ++j) s += a[r][4 + j] * poses[(k + 1) * 16 + j * 4 + c];
      mats[k * 16 + r * 4 + c] = s;
    }
}

// ---- encoder: 3x3x3 conv 18->64, stride (2,2,1), relu, via WMMA ----------
// grid 1250 x 256 threads (8 waves, one z-column tile each). K = 486 pad 512.
__global__ void __launch_bounds__(256) enc_conv_wmma(
    const float* __restrict__ in, const _Float16* __restrict__ w16,
    const float* __restrict__ bias, float* __restrict__ out) {
  __shared__ _Float16 lds[8][16 * 32];
  const int wave = threadIdx.x >> 5, lane = threadIdx.x & 31;
  const int col = blockIdx.x * 8 + wave;
  const int xo = col / kYD, yo = col % kYD;
  _Float16* tile = lds[wave];
  const v8f vz = {0.f,0.f,0.f,0.f,0.f,0.f,0.f,0.f};
  v8f acc0 = vz, acc1 = vz, acc2 = vz, acc3 = vz;
  for (int kc = 0; kc < 16; ++kc) {
    const int kg = kc * 32 + lane;
    const int ci = kg / 27, r = kg % 27;
    const int kx = r / 9, ky = (r / 3) % 3, kz = r % 3;
    const int xin = 2 * xo + kx - 1, yin = 2 * yo + ky - 1;
    const bool okxy = (ci < kNCLS) && ((unsigned)xin < (unsigned)kXF) && ((unsigned)yin < (unsigned)kYF);
#pragma unroll
    for (int m = 0; m < 16; ++m) {
      const int zin = m + kz - 1;
      float v = 0.f;
      if (okxy && (unsigned)zin < 16u)
        v = in[(((size_t)ci * kXF + xin) * kYF + yin) * 16 + zin];
      tile[m * 32 + lane] = (_Float16)v;
    }
    __syncthreads();
    v16h a = frag_a_lds(tile, lane);
    acc0 = WMMA_F16(a, frag_b_glb(w16,  0, 512, kc * 32, lane), acc0);
    acc1 = WMMA_F16(a, frag_b_glb(w16, 16, 512, kc * 32, lane), acc1);
    acc2 = WMMA_F16(a, frag_b_glb(w16, 32, 512, kc * 32, lane), acc2);
    acc3 = WMMA_F16(a, frag_b_glb(w16, 48, 512, kc * 32, lane), acc3);
    __syncthreads();
  }
  const int n = lane & 15, mb = (lane < 16) ? 0 : 8;
  const v8f accs[4] = {acc0, acc1, acc2, acc3};
#pragma unroll
  for (int nt = 0; nt < 4; ++nt) {
    const int co = nt * 16 + n;
    const float bv = bias[co];
    float* dst = out + ((size_t)co * kNCOL + col) * 16 + mb;
#pragma unroll
    for (int rr = 0; rr < 8; ++rr) dst[rr] = fmaxf(accs[nt][rr] + bv, 0.f);
  }
}

// ---- z = slow_feat - fast_feat[0] ----------------------------------------
__global__ void z_init_kernel(const float* __restrict__ slow, const float* __restrict__ fast0,
                              float* __restrict__ z) {
  int i = blockIdx.x * blockDim.x + threadIdx.x;
  if (i >= kCF * kNPOS) return;
  z[i] = slow[i] - fast0[i];
}

// ---- trilinear backward warp of z and fast_feat[k] -----------------------
__global__ void warp_kernel(const float* __restrict__ z, const float* __restrict__ ff,
                            const float* __restrict__ mats, int k,
                            float* __restrict__ zw, float* __restrict__ fadv) {
  int p = blockIdx.x * blockDim.x + threadIdx.x;
  if (p >= kNPOS) return;
  const float* A = mats + k * 16;
  const int zi = p & 15, col = p >> 4;
  const int yo = col % kYD, xo = col / kYD;
  const float wxp = -40.f + (xo + 0.5f) * 0.8f;
  const float wyp = -40.f + (yo + 0.5f) * 0.8f;
  const float wzp = -1.f + (zi + 0.5f) * 0.4f;
  float px = A[0]*wxp + A[1]*wyp + A[2]*wzp + A[3];
  float py = A[4]*wxp + A[5]*wyp + A[6]*wzp + A[7];
  float pz = A[8]*wxp + A[9]*wyp + A[10]*wzp + A[11];
  float sx = fminf(fmaxf((px + 40.f) / 0.8f - 0.5f, 0.f), (float)(kXD - 1));
  float sy = fminf(fmaxf((py + 40.f) / 0.8f - 0.5f, 0.f), (float)(kYD - 1));
  float sz = fminf(fmaxf((pz + 1.f) / 0.4f - 0.5f, 0.f), 15.f);
  float x0 = floorf(sx), y0 = floorf(sy), z0 = floorf(sz);
  float fx = sx - x0, fy = sy - y0, fz = sz - z0;
  int x0i = (int)x0, y0i = (int)y0, z0i = (int)z0;
  int x1i = min(x0i + 1, kXD - 1), y1i = min(y0i + 1, kYD - 1), z1i = min(z0i + 1, 15);
  float w000 = (1-fx)*(1-fy)*(1-fz), w100 = fx*(1-fy)*(1-fz);
  float w010 = (1-fx)*fy*(1-fz),     w110 = fx*fy*(1-fz);
  float w001 = (1-fx)*(1-fy)*fz,     w101 = fx*(1-fy)*fz;
  float w011 = (1-fx)*fy*fz,         w111 = fx*fy*fz;
  int i000 = (x0i*kYD + y0i)*16 + z0i;
  int i100 = (x1i*kYD + y0i)*16 + z0i;
  int i010 = (x0i*kYD + y1i)*16 + z0i;
  int i110 = (x1i*kYD + y1i)*16 + z0i;
  int dz = z1i - z0i;
  for (int c = 0; c < kCF; ++c) {
    const float* b = z + (size_t)c * kNPOS;
    zw[(size_t)c * kNPOS + p] =
        b[i000]*w000 + b[i100]*w100 + b[i010]*w010 + b[i110]*w110 +
        b[i000+dz]*w001 + b[i100+dz]*w101 + b[i010+dz]*w011 + b[i110+dz]*w111;
    const float* f = ff + (size_t)c * kNPOS;
    fadv[(size_t)c * kNPOS + p] =
        f[i000]*w000 + f[i100]*w100 + f[i010]*w010 + f[i110]*w110 +
        f[i000+dz]*w001 + f[i100+dz]*w101 + f[i010+dz]*w011 + f[i110+dz]*w111;
  }
}

// ---- control GEMM: concat(f_t - f_adv, dtau) @ ctrl_w, K=65 pad 96 -------
__global__ void __launch_bounds__(256) ctrl_wmma(
    const float* __restrict__ ft, const float* __restrict__ fadv,
    const int* __restrict__ ts, int kstep,
    const _Float16* __restrict__ w16, const float* __restrict__ bias,
    float* __restrict__ dX) {
  __shared__ _Float16 lds[8][16 * 32];
  const int wave = threadIdx.x >> 5, lane = threadIdx.x & 31;
  const int p0 = (blockIdx.x * 8 + wave) * 16;
  _Float16* tile = lds[wave];
  const float dtau = (float)(ts[kstep + 1] - ts[kstep]) * 1e-6f;
  const v8f vz = {0.f,0.f,0.f,0.f,0.f,0.f,0.f,0.f};
  v8f acc0 = vz, acc1 = vz, acc2 = vz, acc3 = vz;
  for (int kc = 0; kc < 3; ++kc) {
    const int kch = kc * 32 + lane;
#pragma unroll
    for (int m = 0; m < 16; ++m) {
      float v = 0.f;
      if (kch < kCF) v = ft[(size_t)kch * kNPOS + p0 + m] - fadv[(size_t)kch * kNPOS + p0 + m];
      else if (kch == kCF) v = dtau;
      tile[m * 32 + lane] = (_Float16)v;
    }
    __syncthreads();
    v16h a = frag_a_lds(tile, lane);
    acc0 = WMMA_F16(a, frag_b_glb(w16,  0, 96, kc * 32, lane), acc0);
    acc1 = WMMA_F16(a, frag_b_glb(w16, 16, 96, kc * 32, lane), acc1);
    acc2 = WMMA_F16(a, frag_b_glb(w16, 32, 96, kc * 32, lane), acc2);
    acc3 = WMMA_F16(a, frag_b_glb(w16, 48, 96, kc * 32, lane), acc3);
    __syncthreads();
  }
  const int n = lane & 15, mb = (lane < 16) ? 0 : 8;
  const v8f accs[4] = {acc0, acc1, acc2, acc3};
#pragma unroll
  for (int nt = 0; nt < 4; ++nt) {
    const int co = nt * 16 + n;
    const float bv = bias[co];
#pragma unroll
    for (int rr = 0; rr < 8; ++rr)
      dX[(size_t)co * kNPOS + p0 + mb + rr] = accs[nt][rr] + bv;
  }
}

// ---- func_g input GEMM: concat(zw(+k1), fbuf) 128 -> 32, relu ------------
__global__ void __launch_bounds__(256) gin_wmma(
    const float* __restrict__ zw, const float* __restrict__ k1, int addk1,
    const float* __restrict__ fbuf, const _Float16* __restrict__ w16,
    const float* __restrict__ bias, float* __restrict__ h) {
  __shared__ _Float16 lds[8][16 * 32];
  const int wave = threadIdx.x >> 5, lane = threadIdx.x & 31;
  const int p0 = (blockIdx.x * 8 + wave) * 16;
  _Float16* tile = lds[wave];
  const v8f vz = {0.f,0.f,0.f,0.f,0.f,0.f,0.f,0.f};
  v8f acc0 = vz, acc1 = vz;
  for (int kc = 0; kc < 4; ++kc) {
    const int kch = kc * 32 + lane;
#pragma unroll
    for (int m = 0; m < 16; ++m) {
      float v;
      if (kch < kCF) {
        v = zw[(size_t)kch * kNPOS + p0 + m];
        if (addk1) v += k1[(size_t)kch * kNPOS + p0 + m];
      } else {
        v = fbuf[(size_t)(kch - kCF) * kNPOS + p0 + m];
      }
      tile[m * 32 + lane] = (_Float16)v;
    }
    __syncthreads();
    v16h a = frag_a_lds(tile, lane);
    acc0 = WMMA_F16(a, frag_b_glb(w16,  0, 128, kc * 32, lane), acc0);
    acc1 = WMMA_F16(a, frag_b_glb(w16, 16, 128, kc * 32, lane), acc1);
    __syncthreads();
  }
  const int n = lane & 15, mb = (lane < 16) ? 0 : 8;
  const v8f accs[2] = {acc0, acc1};
#pragma unroll
  for (int nt = 0; nt < 2; ++nt) {
    const int co = nt * 16 + n;
    const float bv = bias[co];
#pragma unroll
    for (int rr = 0; rr < 8; ++rr)
      h[(size_t)co * kNPOS + p0 + mb + rr] = fmaxf(accs[nt][rr] + bv, 0.f);
  }
}

// ---- dilated 3x3x3 conv 32->32 (no act), K = 864 -------------------------
__global__ void __launch_bounds__(256) dil_conv_wmma(
    const float* __restrict__ in, const _Float16* __restrict__ w16,
    const float* __restrict__ bias, int d, float* __restrict__ out) {
  __shared__ _Float16 lds[8][16 * 32];
  const int wave = threadIdx.x >> 5, lane = threadIdx.x & 31;
  const int col = blockIdx.x * 8 + wave;
  const int xo = col / kYD, yo = col % kYD;
  _Float16* tile = lds[wave];
  const v8f vz = {0.f,0.f,0.f,0.f,0.f,0.f,0.f,0.f};
  v8f acc0 = vz, acc1 = vz;
  for (int kc = 0; kc < 27; ++kc) {
    const int kg = kc * 32 + lane;
    const int ci = kg / 27, r = kg % 27;
    const int kx = r / 9, ky = (r / 3) % 3, kz = r % 3;
    const int xin = xo + d * (kx - 1), yin = yo + d * (ky - 1);
    const bool ok = ((unsigned)xin < (unsigned)kXD) && ((unsigned)yin < (unsigned)kYD);
    const int dz = d * (kz - 1);
#pragma unroll
    for (int m = 0; m < 16; ++m) {
      const int zin = m + dz;
      float v = 0.f;
      if (ok && (unsigned)zin < 16u)
        v = in[(((size_t)ci * kXD + xin) * kYD + yin) * 16 + zin];
      tile[m * 32 + lane] = (_Float16)v;
    }
    __syncthreads();
    v16h a = frag_a_lds(tile, lane);
    acc0 = WMMA_F16(a, frag_b_glb(w16,  0, 864, kc * 32, lane), acc0);
    acc1 = WMMA_F16(a, frag_b_glb(w16, 16, 864, kc * 32, lane), acc1);
    __syncthreads();
  }
  const int n = lane & 15, mb = (lane < 16) ? 0 : 8;
  const v8f accs[2] = {acc0, acc1};
#pragma unroll
  for (int nt = 0; nt < 2; ++nt) {
    const int co = nt * 16 + n;
    const float bv = bias[co];
    float* dst = out + ((size_t)co * kNCOL + col) * 16 + mb;
#pragma unroll
    for (int rr = 0; rr < 8; ++rr) dst[rr] = accs[nt][rr] + bv;
  }
}

// ---- GroupNorm (8 groups of 4 channels) ----------------------------------
__global__ void gn_stats_kernel(const float* __restrict__ c_buf, float* __restrict__ mv) {
  __shared__ float ssum[256], ssq[256];
  const int g = blockIdx.x, tid = threadIdx.x;
  const float* base = c_buf + (size_t)g * 4 * kNPOS;
  float s = 0.f, q = 0.f;
  for (int i = tid; i < 4 * kNPOS; i += 256) { float v = base[i]; s += v; q += v * v; }
  ssum[tid] = s; ssq[tid] = q;
  __syncthreads();
  for (int st = 128; st > 0; st >>= 1) {
    if (tid < st) { ssum[tid] += ssum[tid + st]; ssq[tid] += ssq[tid + st]; }
    __syncthreads();
  }
  if (tid == 0) {
    const float inv_n = 1.f / (4.f * kNPOS);
    float m = ssum[0] * inv_n;
    float var = ssq[0] * inv_n - m * m;
    mv[g * 2] = m;
    mv[g * 2 + 1] = rsqrtf(var + 1e-5f);
  }
}

__global__ void gn_apply_kernel(const float* __restrict__ c_buf, const float* __restrict__ mv,
                                const float* __restrict__ s, const float* __restrict__ b,
                                float* __restrict__ h) {
  int i = blockIdx.x * blockDim.x + threadIdx.x;
  if (i >= kCI * kNPOS) return;
  int c = i / kNPOS, g = c >> 2;
  float v = (c_buf[i] - mv[2 * g]) * mv[2 * g + 1] * s[c] + b[c];
  h[i] += fmaxf(v, 0.f);
}

// ---- func_g output GEMM: 32 -> 64, tanh, fused *dX and Heun --------------
__global__ void __launch_bounds__(256) gout_wmma(
    const float* __restrict__ h, const _Float16* __restrict__ w16,
    const float* __restrict__ bias, const float* __restrict__ dX,
    const float* __restrict__ zw, const float* __restrict__ k1,
    int mode, float* __restrict__ out) {
  __shared__ _Float16 lds[8][16 * 32];
  const int wave = threadIdx.x >> 5, lane = threadIdx.x & 31;
  const int p0 = (blockIdx.x * 8 + wave) * 16;
  _Float16* tile = lds[wave];
#pragma unroll
  for (int m = 0; m < 16; ++m)
    tile[m * 32 + lane] = (_Float16)h[(size_t)lane * kNPOS + p0 + m];
  __syncthreads();
  const v8f vz = {0.f,0.f,0.f,0.f,0.f,0.f,0.f,0.f};
  v8f acc0 = vz, acc1 = vz, acc2 = vz, acc3 = vz;
  v16h a = frag_a_lds(tile, lane);
  acc0 = WMMA_F16(a, frag_b_glb(w16,  0, 32, 0, lane), acc0);
  acc1 = WMMA_F16(a, frag_b_glb(w16, 16, 32, 0, lane), acc1);
  acc2 = WMMA_F16(a, frag_b_glb(w16, 32, 32, 0, lane), acc2);
  acc3 = WMMA_F16(a, frag_b_glb(w16, 48, 32, 0, lane), acc3);
  const int n = lane & 15, mb = (lane < 16) ? 0 : 8;
  const v8f accs[4] = {acc0, acc1, acc2, acc3};
#pragma unroll
  for (int nt = 0; nt < 4; ++nt) {
    const int co = nt * 16 + n;
    const float bv = bias[co];
#pragma unroll
    for (int rr = 0; rr < 8; ++rr) {
      const size_t idx = (size_t)co * kNPOS + p0 + mb + rr;
      const float g = tanhf(accs[nt][rr] + bv) * dX[idx];
      out[idx] = mode ? (zw[idx] + 0.5f * (k1[idx] + g)) : g;
    }
  }
}

// ---- decoder: nearest x2 upsample fused into 3x3x3 conv 64->18 -----------
// K = 64*27 = 1728, N pad 32; adds dec bias + fast_logits[-1].
__global__ void __launch_bounds__(256) dec_wmma(
    const float* __restrict__ zf, const _Float16* __restrict__ w16,
    const float* __restrict__ bias, const float* __restrict__ fl3,
    float* __restrict__ out) {
  __shared__ _Float16 lds[8][16 * 32];
  const int wave = threadIdx.x >> 5, lane = threadIdx.x & 31;
  const int col = blockIdx.x * 8 + wave;          // 0..39999
  const int xo = col / kYF, yo = col % kYF;
  _Float16* tile = lds[wave];
  const v8f vz = {0.f,0.f,0.f,0.f,0.f,0.f,0.f,0.f};
  v8f acc0 = vz, acc1 = vz;
  for (int kc = 0; kc < 54; ++kc) {
    const int kg = kc * 32 + lane;
    const int ci = kg / 27, r = kg % 27;
    const int kx = r / 9, ky = (r / 3) % 3, kz = r % 3;
    const int xin = xo + kx - 1, yin = yo + ky - 1;
    const bool ok = ((unsigned)xin < (unsigned)kXF) && ((unsigned)yin < (unsigned)kYF);
    const int xs = xin >> 1, ys = yin >> 1;       // nearest upsample source
#pragma unroll
    for (int m = 0; m < 16; ++m) {
      const int zin = m + kz - 1;
      float v = 0.f;
      if (ok && (unsigned)zin < 16u)
        v = zf[(((size_t)ci * kXD + xs) * kYD + ys) * 16 + zin];
      tile[m * 32 + lane] = (_Float16)v;
    }
    __syncthreads();
    v16h a = frag_a_lds(tile, lane);
    acc0 = WMMA_F16(a, frag_b_glb(w16,  0, 1728, kc * 32, lane), acc0);
    acc1 = WMMA_F16(a, frag_b_glb(w16, 16, 1728, kc * 32, lane), acc1);
    __syncthreads();
  }
  const int n = lane & 15, mb = (lane < 16) ? 0 : 8;
  const v8f accs[2] = {acc0, acc1};
#pragma unroll
  for (int nt = 0; nt < 2; ++nt) {
    const int co = nt * 16 + n;
    if (co < kNCLS) {
      const float bv = bias[co];
#pragma unroll
      for (int rr = 0; rr < 8; ++rr) {
        const size_t idx = (((size_t)co * kXF + xo) * kYF + yo) * 16 + mb + rr;
        out[idx] = accs[nt][rr] + bv + fl3[idx];
      }
    }
  }
}

// ---------------------------------------------------------------------------
extern "C" void kernel_launch(void* const* d_in, const int* in_sizes, int n_in,
                              void* d_out, int out_size, void* d_ws, size_t ws_size,
                              hipStream_t stream) {
  (void)in_sizes; (void)n_in; (void)out_size; (void)ws_size;
  const float* fast_logits = (const float*)d_in[0];
  const float* slow_logits = (const float*)d_in[1];
  const float* poses       = (const float*)d_in[2];
  const float* fast_enc_w  = (const float*)d_in[3];
  const float* fast_enc_b  = (const float*)d_in[4];
  const float* slow_enc_w  = (const float*)d_in[5];
  const float* slow_enc_b  = (const float*)d_in[6];
  const float* ctrl_w      = (const float*)d_in[7];
  const float* ctrl_b      = (const float*)d_in[8];
  const float* g_in_w      = (const float*)d_in[9];
  const float* g_in_b      = (const float*)d_in[10];
  const float* g_body_w    = (const float*)d_in[11];
  const float* g_body_b    = (const float*)d_in[12];
  const float* g_gn_s      = (const float*)d_in[13];
  const float* g_gn_b      = (const float*)d_in[14];
  const float* g_out_w     = (const float*)d_in[15];
  const float* g_out_b     = (const float*)d_in[16];
  const float* dec_w       = (const float*)d_in[17];
  const float* dec_b       = (const float*)d_in[18];
  const int*   ts          = (const int*)d_in[19];
  float* out = (float*)d_out;

  char* ws = (char*)d_ws;
  size_t off = 0;
  auto take = [&](size_t bytes) { size_t o = off; off += (bytes + 255) & ~(size_t)255; return o; };

  _Float16* w16_fast = (_Float16*)(ws + take((size_t)64 * 512 * 2));
  _Float16* w16_slow = (_Float16*)(ws + take((size_t)64 * 512 * 2));
  _Float16* w16_ctrl = (_Float16*)(ws + take((size_t)64 * 96 * 2));
  _Float16* w16_gin  = (_Float16*)(ws + take((size_t)32 * 128 * 2));
  _Float16* w16_gb[3];
  for (int i = 0; i < 3; ++i) w16_gb[i] = (_Float16*)(ws + take((size_t)32 * 864 * 2));
  _Float16* w16_gout = (_Float16*)(ws + take((size_t)64 * 32 * 2));
  _Float16* w16_dec  = (_Float16*)(ws + take((size_t)32 * 1728 * 2));
  float* mats  = (float*)(ws + take(3 * 16 * sizeof(float)));
  float* mv    = (float*)(ws + take(16 * sizeof(float)));
  float* fastf = (float*)(ws + take((size_t)4 * kCF * kNPOS * sizeof(float)));
  float* zbuf  = (float*)(ws + take((size_t)kCF * kNPOS * sizeof(float)));
  float* zw    = (float*)(ws + take((size_t)kCF * kNPOS * sizeof(float)));
  float* fadv  = (float*)(ws + take((size_t)kCF * kNPOS * sizeof(float)));
  float* dX    = (float*)(ws + take((size_t)kCF * kNPOS * sizeof(float)));
  float* k1    = (float*)(ws + take((size_t)kCF * kNPOS * sizeof(float)));
  float* hbuf  = (float*)(ws + take((size_t)kCI * kNPOS * sizeof(float)));
  float* cbuf  = (float*)(ws + take((size_t)kCI * kNPOS * sizeof(float)));

  // --- pack weights to f16 [Npad][Kpad] ---
  pack_w_kernel<<<(64*512 + 255)/256, 256, 0, stream>>>(fast_enc_w, w16_fast, 64, 18, 3,3,3, 64, 512);
  pack_w_kernel<<<(64*512 + 255)/256, 256, 0, stream>>>(slow_enc_w, w16_slow, 64, 18, 3,3,3, 64, 512);
  pack_w_kernel<<<(64*96  + 255)/256, 256, 0, stream>>>(ctrl_w,     w16_ctrl, 64, 65, 1,1,1, 64, 96);
  pack_w_kernel<<<(32*128 + 255)/256, 256, 0, stream>>>(g_in_w,     w16_gin,  32, 128,1,1,1, 32, 128);
  for (int i = 0; i < 3; ++i)
    pack_w_kernel<<<(32*864 + 255)/256, 256, 0, stream>>>(g_body_w + (size_t)i*32*32*27, w16_gb[i],
                                                          32, 32, 3,3,3, 32, 864);
  pack_w_kernel<<<(64*32  + 255)/256, 256, 0, stream>>>(g_out_w,    w16_gout, 64, 32, 1,1,1, 64, 32);
  pack_w_kernel<<<(32*1728+ 255)/256, 256, 0, stream>>>(dec_w,      w16_dec,  18, 64, 3,3,3, 32, 1728);

  pose_mats_kernel<<<1, 32, 0, stream>>>(poses, mats);

  // --- encoders (WMMA) ---
  const size_t frame_in = (size_t)kNCLS * kXF * kYF * 16;   // 11,520,000
  const size_t frame_ft = (size_t)kCF * kNPOS;
  for (int t = 0; t < 4; ++t)
    enc_conv_wmma<<<1250, 256, 0, stream>>>(fast_logits + t * frame_in, w16_fast, fast_enc_b,
                                            fastf + t * frame_ft);
  enc_conv_wmma<<<1250, 256, 0, stream>>>(slow_logits, w16_slow, slow_enc_b, zbuf);
  z_init_kernel<<<(kCF * kNPOS + 255)/256, 256, 0, stream>>>(zbuf, fastf, zbuf);

  // --- Heun steps ---
  for (int k = 0; k < 3; ++k) {
    warp_kernel<<<(kNPOS + 255)/256, 256, 0, stream>>>(zbuf, fastf + (size_t)k * frame_ft, mats, k, zw, fadv);
    ctrl_wmma<<<1250, 256, 0, stream>>>(fastf + (size_t)(k + 1) * frame_ft, fadv, ts, k,
                                        w16_ctrl, ctrl_b, dX);
    // func_g #1: (zw, fadv) -> k1 = tanh(...)*dX
    gin_wmma<<<1250, 256, 0, stream>>>(zw, nullptr, 0, fadv, w16_gin, g_in_b, hbuf);
    for (int i = 0; i < 3; ++i) {
      dil_conv_wmma<<<1250, 256, 0, stream>>>(hbuf, w16_gb[i], g_body_b + i*32, i + 1, cbuf);
      gn_stats_kernel<<<8, 256, 0, stream>>>(cbuf, mv);
      gn_apply_kernel<<<(kCI * kNPOS + 255)/256, 256, 0, stream>>>(cbuf, mv, g_gn_s + i*32,
                                                                   g_gn_b + i*32, hbuf);
    }
    gout_wmma<<<1250, 256, 0, stream>>>(hbuf, w16_gout, g_out_b, dX, zw, k1, 0, k1);
    // func_g #2: (zw + k1, f_t) -> z = zw + 0.5*(k1 + tanh(...)*dX)
    gin_wmma<<<1250, 256, 0, stream>>>(zw, k1, 1, fastf + (size_t)(k + 1) * frame_ft,
                                       w16_gin, g_in_b, hbuf);
    for (int i = 0; i < 3; ++i) {
      dil_conv_wmma<<<1250, 256, 0, stream>>>(hbuf, w16_gb[i], g_body_b + i*32, i + 1, cbuf);
      gn_stats_kernel<<<8, 256, 0, stream>>>(cbuf, mv);
      gn_apply_kernel<<<(kCI * kNPOS + 255)/256, 256, 0, stream>>>(cbuf, mv, g_gn_s + i*32,
                                                                   g_gn_b + i*32, hbuf);
    }
    gout_wmma<<<1250, 256, 0, stream>>>(hbuf, w16_gout, g_out_b, dX, zw, k1, 1, zbuf);
  }

  // --- decoder + residual ---
  dec_wmma<<<5000, 256, 0, stream>>>(zbuf, w16_dec, dec_b, fast_logits + 3 * frame_in, out);
}
